// SlotAttentionMulti_40553081209572
// MI455X (gfx1250) — matrix operand order repaired and checked by reading, not compiled, not run on hardware
//
#include <hip/hip_runtime.h>
#include <hip/hip_bf16.h>

typedef __attribute__((ext_vector_type(16))) __bf16         v16bf;
typedef __attribute__((ext_vector_type(8)))  float          v8f;
typedef __attribute__((ext_vector_type(8)))  unsigned short v8u16;

#define DEV static __device__ __forceinline__

// ---- problem constants ----
#define B_   16
#define V_   4
#define BV   64
#define N_   4096
#define FIN  256
#define QS   128
#define SV   64
#define SA   192
#define SF   256
#define FR   512
#define S_   8
#define ROWS (BV * S_)      // 512 slot rows
#define NSTEPS 3

DEV unsigned lane_id() { return threadIdx.x & 31u; }

DEV unsigned short f2bf(float f) {
  union { float f; unsigned int u; } c; c.f = f;
  unsigned int u = c.u;
  u += 0x7FFFu + ((u >> 16) & 1u);          // round to nearest even
  return (unsigned short)(u >> 16);
}

// A fragment: 16x32 bf16, A row-major (ld in elements). ISA layout:
// lane<16: row=lane, k in {0..7, 16..23}; lane>=16: row=lane-16, k in {8..15, 24..31}
DEV v16bf fragA(const unsigned short* A, size_t row0, int k0, int ld) {
  unsigned l = lane_id();
  const unsigned short* p = A + (row0 + (l & 15u)) * (size_t)ld + (unsigned)k0 + ((l >> 4u) << 3u);
  union { v8u16 h[2]; v16bf v; } u;
  u.h[0] = *(const v8u16*)p;
  u.h[1] = *(const v8u16*)(p + 16);
  return u.v;
}

// B fragment: 32x16 bf16 from B^T stored row-major N x K.
// lane L: col n = L&15; k in {8h..8h+7, 16+8h..16+8h+7}, h = L>>4.
DEV v16bf fragB(const unsigned short* Bt, size_t n0, int k0, int ld) {
  unsigned l = lane_id();
  const unsigned short* p = Bt + (n0 + (l & 15u)) * (size_t)ld + (unsigned)k0 + ((l >> 4u) << 3u);
  union { v8u16 h[2]; v16bf v; } u;
  u.h[0] = *(const v8u16*)p;
  u.h[1] = *(const v8u16*)(p + 16);
  return u.v;
}

DEV v8f wmma_bf16(v16bf a, v16bf b, v8f c) {
  return __builtin_amdgcn_wmma_f32_16x16x32_bf16(false, a, false, b, (short)0, c, false, false);
}

// ------------------------------------------------------------------
// Fused LN + kv projection: rows = BV*N_, K=FIN, cols 0..127 -> xkey*coef,
// cols 128..383 -> xvalT (transposed per batch: [b][f][n], bf16)
// ------------------------------------------------------------------
__global__ __launch_bounds__(256)
void kv_proj_kernel(const float* __restrict__ x,
                    const float* __restrict__ g, const float* __restrict__ bta,
                    const unsigned short* __restrict__ WkvT,
                    unsigned short* __restrict__ xkey,
                    unsigned short* __restrict__ xvalT,
                    float coef)
{
  __shared__ float xs[16][FIN];
  __shared__ __attribute__((aligned(16))) unsigned short as[16][FIN + 8];
  const int tid = threadIdx.x;
  const size_t row0 = (size_t)blockIdx.x * 16;

  // cooperative load of 16x256 f32 tile (1024 float4)
  #pragma unroll
  for (int it = 0; it < 4; ++it) {
    int j = tid + it * 256;
    int r = j >> 6;
    int c = (j & 63) << 2;
    float4 v = *(const float4*)(x + (row0 + r) * FIN + c);
    xs[r][c] = v.x; xs[r][c + 1] = v.y; xs[r][c + 2] = v.z; xs[r][c + 3] = v.w;
  }
  if (blockIdx.x + 1 < gridDim.x)
    __builtin_prefetch(x + (row0 + 16) * FIN + (tid << 4), 0, 1);
  __syncthreads();

  // LayerNorm: wave w normalizes rows 2w, 2w+1 (wave32 shuffle reduction)
  const int wave = tid >> 5;
  const unsigned l = lane_id();
  #pragma unroll
  for (int rr = 0; rr < 2; ++rr) {
    int r = wave * 2 + rr;
    float s = 0.f, s2 = 0.f;
    #pragma unroll
    for (int j = 0; j < FIN / 32; ++j) {
      float v = xs[r][l + 32 * j];
      s += v; s2 += v * v;
    }
    #pragma unroll
    for (int off = 16; off > 0; off >>= 1) {
      s  += __shfl_xor(s,  off, 32);
      s2 += __shfl_xor(s2, off, 32);
    }
    float mu   = s * (1.f / FIN);
    float var  = s2 * (1.f / FIN) - mu * mu;
    float rstd = rsqrtf(var + 1e-5f);
    #pragma unroll
    for (int j = 0; j < FIN / 32; ++j) {
      int c = l + 32 * j;
      as[r][c] = f2bf((xs[r][c] - mu) * rstd * g[c] + bta[c]);
    }
  }
  __syncthreads();

  // each wave: 3 column tiles {w, w+8, w+16} of 24 (=384/16)
  v8f acc0 = {}; v8f acc1 = {}; v8f acc2 = {};
  for (int k0 = 0; k0 < FIN; k0 += 32) {
    union { v8u16 h[2]; v16bf v; } ua;
    {
      const unsigned short* p = &as[l & 15u][(unsigned)k0 + ((l >> 4u) << 3u)];
      ua.h[0] = *(const v8u16*)p;          // LDS 128-bit loads
      ua.h[1] = *(const v8u16*)(p + 16);
    }
    v16bf b0 = fragB(WkvT, (size_t)(wave)      * 16, k0, FIN);
    v16bf b1 = fragB(WkvT, (size_t)(wave + 8)  * 16, k0, FIN);
    v16bf b2 = fragB(WkvT, (size_t)(wave + 16) * 16, k0, FIN);
    acc0 = wmma_bf16(ua.v, b0, acc0);
    acc1 = wmma_bf16(ua.v, b1, acc1);
    acc2 = wmma_bf16(ua.v, b2, acc2);
  }

  const int col = l & 15;
  const int rb  = (l >> 4) * 8;
  #pragma unroll
  for (int j = 0; j < 3; ++j) {
    int gcol = (wave + 8 * j) * 16 + col;
    v8f a = (j == 0) ? acc0 : ((j == 1) ? acc1 : acc2);
    #pragma unroll
    for (int i = 0; i < 8; ++i) {
      size_t grow = row0 + rb + i;
      float v = a[i];
      if (gcol < QS) {
        xkey[grow * QS + gcol] = f2bf(v * coef);
      } else {
        size_t bb = grow >> 12;                   // / N_
        size_t n  = grow & (N_ - 1);
        xvalT[(bb * SF + (size_t)(gcol - QS)) * N_ + n] = f2bf(v);
      }
    }
  }
}

// ------------------------------------------------------------------
// Generic bf16 GEMM: C[m,n] = scale * sum_k A[m,k]*Bt[n,k] (+ bias[n])
// mode bit0: bf16 output, bit1: relu, bit2: remap rows 8->16 padding
// ------------------------------------------------------------------
__global__ __launch_bounds__(256)
void gemm_bf16_kernel(const unsigned short* __restrict__ A, int lda,
                      const unsigned short* __restrict__ Bt, int ldb,
                      float* __restrict__ Cf, unsigned short* __restrict__ Cb, int ldc,
                      int tilesM, int tilesN, int K, int Mvalid,
                      float scale, const float* __restrict__ bias, int mode,
                      long long strideA, long long strideB, long long strideC)
{
  const int wave = threadIdx.x >> 5;
  const int t = blockIdx.x * 8 + wave;
  if (t >= tilesM * tilesN) return;
  const int tm = t / tilesN, tn = t % tilesN;
  const long long bz = blockIdx.z;
  A  += bz * strideA;
  Bt += bz * strideB;

  v8f acc = {};
  for (int k0 = 0; k0 < K; k0 += 32) {
    v16bf a = fragA(A, (size_t)tm * 16, k0, lda);
    v16bf b = fragB(Bt, (size_t)tn * 16, k0, ldb);
    acc = wmma_bf16(a, b, acc);
  }

  const unsigned l = lane_id();
  const int col = l & 15;
  const int rb  = (l >> 4) * 8;
  const int gcol = tn * 16 + col;
  const float bv = bias ? bias[gcol] : 0.f;
  #pragma unroll
  for (int i = 0; i < 8; ++i) {
    int grow = tm * 16 + rb + i;
    if (grow >= Mvalid) continue;
    float v = acc[i] * scale + bv;
    if (mode & 2) v = fmaxf(v, 0.f);
    long long orow = grow;
    if (mode & 4) orow = ((long long)(grow >> 3) << 4) | (grow & 7);
    if (mode & 1) Cb[bz * strideC + orow * (long long)ldc + gcol] = f2bf(v);
    else          Cf[bz * strideC + orow * (long long)ldc + gcol] = v;
  }
}

// ---- LayerNorm over 256 cols, bf16 output; one wave per row ----
__global__ __launch_bounds__(256)
void ln256_bf16_kernel(const float* __restrict__ in, const float* __restrict__ g,
                       const float* __restrict__ bb, unsigned short* __restrict__ out,
                       int rows)
{
  int row = blockIdx.x * 8 + (threadIdx.x >> 5);
  if (row >= rows) return;
  const unsigned l = lane_id();
  const float* p = in + (size_t)row * 256;
  float v[8], s = 0.f, s2 = 0.f;
  #pragma unroll
  for (int j = 0; j < 8; ++j) { v[j] = p[l + 32 * j]; s += v[j]; s2 += v[j] * v[j]; }
  #pragma unroll
  for (int off = 16; off > 0; off >>= 1) {
    s += __shfl_xor(s, off, 32); s2 += __shfl_xor(s2, off, 32);
  }
  float mu = s * (1.f / 256.f), var = s2 * (1.f / 256.f) - mu * mu;
  float rstd = rsqrtf(var + 1e-5f);
  unsigned short* q = out + (size_t)row * 256;
  #pragma unroll
  for (int j = 0; j < 8; ++j) {
    int c = l + 32 * j;
    q[c] = f2bf((v[j] - mu) * rstd * g[c] + bb[c]);
  }
}

// ---- small elementwise kernels ----
__global__ void conv_bf16_kernel(const float* in, unsigned short* out, int n) {
  int i = blockIdx.x * 256 + threadIdx.x;
  if (i < n) out[i] = f2bf(in[i]);
}
__global__ void convT_bf16_kernel(const float* in, unsigned short* out, int R, int C) {
  int i = blockIdx.x * 256 + threadIdx.x;
  if (i < R * C) { int r = i / C, c = i % C; out[(size_t)c * R + r] = f2bf(in[i]); }
}
__global__ void zero_u16_kernel(unsigned short* p, int n) {
  int i = blockIdx.x * 256 + threadIdx.x;
  if (i < n) p[i] = 0;
}
__global__ void init_slots_kernel(const float* nv, const float* na,
    const float* vloc, const float* vls, const float* aloc, const float* als,
    float* slot_view, float* slot_attr)
{
  int idx = blockIdx.x * 256 + threadIdx.x;      // 4096 + 24576 threads
  if (idx < BV * SV) {
    int c = idx & (SV - 1);
    slot_view[idx] = vloc[c] + expf(vls[c]) * nv[idx];
  } else {
    int j = idx - BV * SV;
    int c = j % SA;
    slot_attr[j] = aloc[c] + expf(als[c]) * na[j];
  }
}
__global__ void build_slot_kernel(const float* sv, const float* sa,
                                  float* slot_full, unsigned short* h_bf)
{
  int idx = blockIdx.x * 256 + threadIdx.x;      // ROWS*SF
  int row = idx >> 8, c = idx & 255;
  int bv = row >> 3, s = row & 7, b = bv >> 2;
  float v = (c < SV) ? sv[bv * SV + c] : sa[(b * S_ + s) * SA + (c - SV)];
  slot_full[idx] = v;
  h_bf[idx] = f2bf(v);
}
__global__ void col_lsm_kernel(float* logits)     // log_softmax over S per (b,n)
{
  int idx = blockIdx.x * 256 + threadIdx.x;
  if (idx >= BV * N_) return;
  int b = idx >> 12, n = idx & (N_ - 1);
  float* p = logits + (size_t)b * S_ * N_ + n;
  float v[S_], m = -3.4e38f;
  #pragma unroll
  for (int s = 0; s < S_; ++s) { v[s] = p[(size_t)s * N_]; m = fmaxf(m, v[s]); }
  float sum = 0.f;
  #pragma unroll
  for (int s = 0; s < S_; ++s) sum += expf(v[s] - m);
  float lse = m + logf(sum);
  #pragma unroll
  for (int s = 0; s < S_; ++s) p[(size_t)s * N_] = v[s] - lse;
}
__global__ __launch_bounds__(256)
void row_lse_kernel(const float* __restrict__ logits, float* __restrict__ rowlse)
{
  __shared__ float red[256];
  const float* p = logits + (size_t)blockIdx.x * N_;
  int tid = threadIdx.x;
  float m = -3.4e38f;
  for (int i = tid; i < N_; i += 256) m = fmaxf(m, p[i]);
  red[tid] = m; __syncthreads();
  for (int off = 128; off > 0; off >>= 1) {
    if (tid < off) red[tid] = fmaxf(red[tid], red[tid + off]);
    __syncthreads();
  }
  m = red[0]; __syncthreads();
  float s = 0.f;
  for (int i = tid; i < N_; i += 256) s += expf(p[i] - m);
  red[tid] = s; __syncthreads();
  for (int off = 128; off > 0; off >>= 1) {
    if (tid < off) red[tid] += red[tid + off];
    __syncthreads();
  }
  if (tid == 0) rowlse[blockIdx.x] = m + logf(red[0]);
}
__global__ void attn_bf16_kernel(const float* logits, const float* rowlse,
                                 unsigned short* attn)
{
  int idx = blockIdx.x * 256 + threadIdx.x;       // BV*16*N_
  if (idx >= BV * 16 * N_) return;
  int n = idx & (N_ - 1);
  int r = (idx >> 12) & 15;
  int b = idx >> 16;
  unsigned short o = 0;
  if (r < S_) o = f2bf(expf(logits[((size_t)b * S_ + r) * N_ + n] - rowlse[b * S_ + r]));
  attn[idx] = o;
}
__global__ void gru_gate_kernel(const float* gi, const float* gh,
                                const float* slot_full, float* x_main)
{
  int idx = blockIdx.x * 256 + threadIdx.x;       // ROWS*SF
  int row = idx >> 8, c = idx & 255;
  const float* pi = gi + (size_t)row * 768;
  const float* ph = gh + (size_t)row * 768;
  float r  = 1.f / (1.f + expf(-(pi[c]       + ph[c])));
  float z  = 1.f / (1.f + expf(-(pi[256 + c] + ph[256 + c])));
  float nn = tanhf(pi[512 + c] + r * ph[512 + c]);
  float h  = slot_full[idx];
  x_main[idx] = (1.f - z) * nn + z * h;
}
__global__ void add_res_kernel(const float* x_main, const float* res, float* slot_full) {
  int idx = blockIdx.x * 256 + threadIdx.x;
  slot_full[idx] = x_main[idx] + res[idx];
}
__global__ void reduce_view_kernel(const float* slot_full, float* slot_view) {
  int idx = blockIdx.x * 256 + threadIdx.x;       // BV*SV
  if (idx >= BV * SV) return;
  int bv = idx >> 6, c = idx & 63;
  float s = 0.f;
  #pragma unroll
  for (int ss = 0; ss < S_; ++ss) s += slot_full[((size_t)(bv * S_ + ss)) * SF + c];
  slot_view[idx] = s * (1.f / S_);
}
__global__ void reduce_attr_kernel(const float* slot_full, float* slot_attr) {
  int idx = blockIdx.x * 256 + threadIdx.x;       // B_*S_*SA
  if (idx >= B_ * S_ * SA) return;
  int c = idx % SA; int s = (idx / SA) & 7; int b = idx / (SA * S_);
  float acc = 0.f;
  #pragma unroll
  for (int v = 0; v < V_; ++v)
    acc += slot_full[((size_t)((b * V_ + v) * S_ + s)) * SF + SV + c];
  slot_attr[idx] = acc * (1.f / V_);
}
__global__ void write_out_kernel(const float* sv, const float* sa, float* out) {
  int i = blockIdx.x * 256 + threadIdx.x;         // 28672
  if (i < BV * SV) out[i] = sv[i];
  else             out[i] = sa[i - BV * SV];
}

// ------------------------------------------------------------------
extern "C" void kernel_launch(void* const* d_in, const int* in_sizes, int n_in,
                              void* d_out, int out_size, void* d_ws, size_t ws_size,
                              hipStream_t stream)
{
  (void)in_sizes; (void)n_in; (void)out_size; (void)ws_size;
  const float* x       = (const float*)d_in[0];
  const float* noise_v = (const float*)d_in[1];
  const float* noise_a = (const float*)d_in[2];
  const float* vloc    = (const float*)d_in[3];
  const float* vls     = (const float*)d_in[4];
  const float* aloc    = (const float*)d_in[5];
  const float* als     = (const float*)d_in[6];
  const float* ln_kv_g = (const float*)d_in[7];
  const float* ln_kv_b = (const float*)d_in[8];
  const float* W_kv    = (const float*)d_in[9];
  const float* ln_q_g  = (const float*)d_in[10];
  const float* ln_q_b  = (const float*)d_in[11];
  const float* W_qry   = (const float*)d_in[12];
  const float* w_ih    = (const float*)d_in[13];
  const float* w_hh    = (const float*)d_in[14];
  const float* b_ih    = (const float*)d_in[15];
  const float* b_hh    = (const float*)d_in[16];
  const float* ln_r_g  = (const float*)d_in[17];
  const float* ln_r_b  = (const float*)d_in[18];
  const float* W_r1    = (const float*)d_in[19];
  const float* b_r1    = (const float*)d_in[20];
  const float* W_r2    = (const float*)d_in[21];
  const float* b_r2    = (const float*)d_in[22];

  char* ws = (char*)d_ws;
  size_t off = 0;
  auto take = [&](size_t bytes) -> char* {
    char* p = ws + off;
    off += (bytes + 255) & ~(size_t)255;
    return p;
  };
  unsigned short* xkey     = (unsigned short*)take((size_t)BV * N_ * QS * 2);
  unsigned short* xvalT    = (unsigned short*)take((size_t)BV * SF * N_ * 2);
  unsigned short* WkvT     = (unsigned short*)take((size_t)384 * 256 * 2);
  unsigned short* WqryT    = (unsigned short*)take((size_t)128 * 256 * 2);
  unsigned short* wih_bf   = (unsigned short*)take((size_t)768 * 256 * 2);
  unsigned short* whh_bf   = (unsigned short*)take((size_t)768 * 256 * 2);
  unsigned short* Wr1T     = (unsigned short*)take((size_t)512 * 256 * 2);
  unsigned short* Wr2T     = (unsigned short*)take((size_t)256 * 512 * 2);
  float*          slot_view= (float*)take((size_t)BV * SV * 4);
  float*          slot_attr= (float*)take((size_t)B_ * S_ * SA * 4);
  float*          slot_full= (float*)take((size_t)ROWS * SF * 4);
  unsigned short* h_bf     = (unsigned short*)take((size_t)ROWS * SF * 2);
  unsigned short* slotq_ln = (unsigned short*)take((size_t)ROWS * SF * 2);
  unsigned short* xqry_pad = (unsigned short*)take((size_t)BV * 16 * QS * 2);
  float*          logits   = (float*)take((size_t)BV * S_ * N_ * 4);
  float*          rowlse   = (float*)take((size_t)ROWS * 4);
  unsigned short* attn_bf  = (unsigned short*)take((size_t)BV * 16 * N_ * 2);
  unsigned short* xupd_bf  = (unsigned short*)take((size_t)ROWS * SF * 2);
  float*          gi       = (float*)take((size_t)ROWS * 768 * 4);
  float*          gh       = (float*)take((size_t)ROWS * 768 * 4);
  float*          x_main   = (float*)take((size_t)ROWS * SF * 4);
  unsigned short* xmain_ln = (unsigned short*)take((size_t)ROWS * SF * 2);
  unsigned short* h1_bf    = (unsigned short*)take((size_t)ROWS * FR * 2);
  float*          res      = (float*)take((size_t)ROWS * SF * 4);

  const float coef = 0.08838834764831845f;   // 1/sqrt(128)

  // weight conversions (bf16, B^T layout)
  convT_bf16_kernel<<<(256 * 384 + 255) / 256, 256, 0, stream>>>(W_kv,  WkvT, 256, 384);
  convT_bf16_kernel<<<(256 * 128 + 255) / 256, 256, 0, stream>>>(W_qry, WqryT, 256, 128);
  conv_bf16_kernel <<<(768 * 256 + 255) / 256, 256, 0, stream>>>(w_ih, wih_bf, 768 * 256);
  conv_bf16_kernel <<<(768 * 256 + 255) / 256, 256, 0, stream>>>(w_hh, whh_bf, 768 * 256);
  convT_bf16_kernel<<<(256 * 512 + 255) / 256, 256, 0, stream>>>(W_r1, Wr1T, 256, 512);
  convT_bf16_kernel<<<(512 * 256 + 255) / 256, 256, 0, stream>>>(W_r2, Wr2T, 512, 256);

  init_slots_kernel<<<112, 256, 0, stream>>>(noise_v, noise_a, vloc, vls, aloc, als,
                                             slot_view, slot_attr);

  // fused LN + kv projection (dominant GEMM, WMMA bf16)
  kv_proj_kernel<<<(BV * N_) / 16, 256, 0, stream>>>(x, ln_kv_g, ln_kv_b, WkvT,
                                                     xkey, xvalT, coef);

  for (int step = 0; step < NSTEPS; ++step) {
    build_slot_kernel<<<ROWS, 256, 0, stream>>>(slot_view, slot_attr, slot_full, h_bf);
    ln256_bf16_kernel<<<ROWS / 8, 256, 0, stream>>>(slot_full, ln_q_g, ln_q_b, slotq_ln, ROWS);
    zero_u16_kernel<<<(BV * 16 * QS + 255) / 256, 256, 0, stream>>>(xqry_pad, BV * 16 * QS);

    // x_qry = LN(slot_full) @ W_qry -> bf16, rows remapped 8->16 padding
    gemm_bf16_kernel<<<dim3(32, 1, 1), 256, 0, stream>>>(
        slotq_ln, SF, WqryT, SF, nullptr, xqry_pad, QS,
        ROWS / 16, QS / 16, SF, ROWS, 1.f, nullptr, 1 | 4, 0, 0, 0);

    // logits = qry @ key^T, batched over 64 (b,v)
    gemm_bf16_kernel<<<dim3(32, 1, BV), 256, 0, stream>>>(
        xqry_pad, QS, xkey, QS, logits, nullptr, N_,
        1, N_ / 16, QS, S_, 1.f, nullptr, 0,
        (long long)16 * QS, (long long)N_ * QS, (long long)S_ * N_);

    col_lsm_kernel<<<(BV * N_ + 255) / 256, 256, 0, stream>>>(logits);
    row_lse_kernel<<<ROWS, 256, 0, stream>>>(logits, rowlse);
    attn_bf16_kernel<<<(BV * 16 * N_ + 255) / 256, 256, 0, stream>>>(logits, rowlse, attn_bf);

    // x_upd = attn @ x_val, batched, bf16 out
    gemm_bf16_kernel<<<dim3(2, 1, BV), 256, 0, stream>>>(
        attn_bf, N_, xvalT, N_, nullptr, xupd_bf, SF,
        1, SF / 16, N_, S_, 1.f, nullptr, 1,
        (long long)16 * N_, (long long)SF * N_, (long long)S_ * SF);

    // GRU gates
    gemm_bf16_kernel<<<dim3(192, 1, 1), 256, 0, stream>>>(
        xupd_bf, SF, wih_bf, SF, gi, nullptr, 768,
        ROWS / 16, 768 / 16, SF, ROWS, 1.f, b_ih, 0, 0, 0, 0);
    gemm_bf16_kernel<<<dim3(192, 1, 1), 256, 0, stream>>>(
        h_bf, SF, whh_bf, SF, gh, nullptr, 768,
        ROWS / 16, 768 / 16, SF, ROWS, 1.f, b_hh, 0, 0, 0, 0);
    gru_gate_kernel<<<ROWS, 256, 0, stream>>>(gi, gh, slot_full, x_main);

    // residual MLP
    ln256_bf16_kernel<<<ROWS / 8, 256, 0, stream>>>(x_main, ln_r_g, ln_r_b, xmain_ln, ROWS);
    gemm_bf16_kernel<<<dim3(128, 1, 1), 256, 0, stream>>>(
        xmain_ln, SF, Wr1T, SF, nullptr, h1_bf, FR,
        ROWS / 16, FR / 16, SF, ROWS, 1.f, b_r1, 1 | 2, 0, 0, 0);
    gemm_bf16_kernel<<<dim3(64, 1, 1), 256, 0, stream>>>(
        h1_bf, FR, Wr2T, FR, res, nullptr, SF,
        ROWS / 16, SF / 16, FR, ROWS, 1.f, b_r2, 0, 0, 0, 0);

    add_res_kernel<<<ROWS, 256, 0, stream>>>(x_main, res, slot_full);
    reduce_view_kernel<<<16, 256, 0, stream>>>(slot_full, slot_view);
    reduce_attr_kernel<<<96, 256, 0, stream>>>(slot_full, slot_attr);
  }

  write_out_kernel<<<112, 256, 0, stream>>>(slot_view, slot_attr, (float*)d_out);
}
